// TriangleMultiplicativeModule_14216341750246
// MI455X (gfx1250) — matrix operand order repaired and checked by reading, not compiled, not run on hardware
//
#include <hip/hip_runtime.h>
#include <hip/hip_bf16.h>

// ---------------------------------------------------------------------------
// TriangleMultiplicativeModule (ingoing) for MI455X / gfx1250, wave32 + WMMA.
// k_cvt (weights->bf16) -> k_proj (LN + 5 gated projections, branch-free WMMA)
// -> k_tri (per-channel 384^3 GEMM, TDM tensor_load_to_lds double buffering)
// -> k_out (LN + output projection + gate).  Workspace ~227 MB.
// ---------------------------------------------------------------------------

typedef __bf16 bf16_t;
typedef __attribute__((ext_vector_type(8)))  bf16_t bf16x8;
typedef __attribute__((ext_vector_type(16))) bf16_t bf16x16;
typedef __attribute__((ext_vector_type(8)))  float  f32x8;
typedef __attribute__((ext_vector_type(4)))  unsigned int u32x4;
typedef __attribute__((ext_vector_type(8)))  int i32x8;
typedef __attribute__((ext_vector_type(4)))  int i32x4;

#define SEQN 384
#define DIMD 128
#define HIDH 128
#define NN   (SEQN * SEQN)
#define LNEPS 1e-5f

static __device__ __forceinline__ bf16_t to_bf16(float x) { return (bf16_t)x; }
static __device__ __forceinline__ float sigm(float x) { return 1.0f / (1.0f + __expf(-x)); }

static __device__ __forceinline__ f32x8 wmma_bf16(bf16x16 a, bf16x16 b, f32x8 c) {
  return __builtin_amdgcn_wmma_f32_16x16x32_bf16(false, a, false, b, (short)0, c,
                                                 false, false);
}

// A-operand (16x32, 16-bit): lane holds row M=lane%16; lanes 0-15 carry
// K = kb+[0..7] and kb+[16..23]; lanes 16-31 carry kb+[8..15] and kb+[24..31].
static __device__ __forceinline__ bf16x16 load_a(const bf16_t* tile, int ld,
                                                 int lane, int kb) {
  const int row = lane & 15;
  const int sel = (lane < 16) ? 0 : 8;
  bf16x8 lo = *(const bf16x8*)(tile + (size_t)row * ld + kb + sel);
  bf16x8 hi = *(const bf16x8*)(tile + (size_t)row * ld + kb + 16 + sel);
  bf16x16 a;
#pragma unroll
  for (int e = 0; e < 8; ++e) { a[e] = lo[e]; a[e + 8] = hi[e]; }
  return a;
}

// B-operand (32x16, 16-bit) from B^T stored row-major [n][k]: lane holds
// column n=lane%16; lanes 0-15 carry K=kb+[0..7] and kb+[16..23]; lanes 16-31
// carry kb+[8..15] and kb+[24..31].
static __device__ __forceinline__ bf16x16 load_bt(const bf16_t* bt, int ld,
                                                  int lane, int kb) {
  const int row = lane & 15;
  const int sel = (lane < 16) ? 0 : 8;
  bf16x8 lo = *(const bf16x8*)(bt + (size_t)row * ld + kb + sel);
  bf16x8 hi = *(const bf16x8*)(bt + (size_t)row * ld + kb + 16 + sel);
  bf16x16 b;
#pragma unroll
  for (int e = 0; e < 8; ++e) { b[e] = lo[e]; b[e + 8] = hi[e]; }
  return b;
}

// --- Tensor Data Mover: 2D tile (rows x 32 bf16 elements) global -> LDS ----
static __device__ __forceinline__ unsigned lds_off(const void* p) {
  // Generic LDS address truncated to 32 bits == LDS offset (ISA flat rules).
  return (unsigned)(uintptr_t)p;
}

static __device__ __forceinline__ void tdm_load_2d(unsigned ldsa,
                                                   const void* gaddr,
                                                   int rows) {
  const unsigned long long ga = (unsigned long long)(uintptr_t)gaddr;
  u32x4 g0;
  g0.x = 1u;                                     // count=1 user descriptor
  g0.y = ldsa;                                   // lds_addr (bytes)
  g0.z = (unsigned)ga;                           // global_addr[31:0]
  g0.w = (unsigned)((ga >> 32) & 0x01FFFFFFu) |  // global_addr[56:32]
         (2u << 30);                             // type=2 ("image")
  i32x8 g1;
  g1[0] = 0x00010000;                 // workgroup_mask=0, data_size=1 (2B)
  g1[1] = (int)(32u << 16);           // tensor_dim0 = 32  (bits 63:48)
  g1[2] = (int)((unsigned)rows << 16);// tensor_dim1 = rows (bits 95:80)
  g1[3] = (int)(32u << 16);           // tile_dim0 = 32    (bits 127:112)
  g1[4] = rows;                       // tile_dim1 = rows  (bits 143:128)
  g1[5] = SEQN;                       // tensor_dim0_stride = 384 (bits 191:160)
  g1[6] = 0;                          // stride hi / dim1_stride lo
  g1[7] = 0;
  const i32x4 z4 = {0, 0, 0, 0};
#if defined(__clang_major__) && (__clang_major__ >= 23)
  const i32x8 z8 = {0, 0, 0, 0, 0, 0, 0, 0};
  __builtin_amdgcn_tensor_load_to_lds(g0, g1, z4, z4, z8, 0);
#else
  __builtin_amdgcn_tensor_load_to_lds(g0, g1, z4, z4, 0);
#endif
}

// ---------------------------------------------------------------------------
// Kernel 0: convert the six 128x128 fp32 weight matrices to bf16.
// Order in dst: left, right, lgate, rgate, ogate, out_w.
// ---------------------------------------------------------------------------
__global__ __launch_bounds__(256) void k_cvt(const float* __restrict__ s0,
                                             const float* __restrict__ s1,
                                             const float* __restrict__ s2,
                                             const float* __restrict__ s3,
                                             const float* __restrict__ s4,
                                             const float* __restrict__ s5,
                                             bf16_t* __restrict__ dst) {
  const int mat = blockIdx.y;
  const float* s = (mat == 0) ? s0 : (mat == 1) ? s1 : (mat == 2) ? s2
                 : (mat == 3) ? s3 : (mat == 4) ? s4 : s5;
  const int i = blockIdx.x * 256 + threadIdx.x;   // 0..16383
  dst[(size_t)mat * 16384 + i] = to_bf16(s[i]);
}

// ---------------------------------------------------------------------------
// Kernel 1: LayerNorm + five gated projections for a 16-point tile.
// 256 threads = 8 waves; wave w owns h-tile [16w,16w+16) and runs three
// branch-free phases: left*sig(lgate), right*sig(rgate), sigmoid(ogate).
// Outputs: Lws[h][j][k], Rws[h][i][k] (bf16, pre-transposed for k_tri),
// Gws[m][h] (fp32 sigmoid(ogate)).
// ---------------------------------------------------------------------------
__global__ __launch_bounds__(256) void k_proj(
    const float* __restrict__ x, const float* __restrict__ nw,
    const float* __restrict__ nb, const bf16_t* __restrict__ wB,
    const float* __restrict__ bL, const float* __restrict__ bR,
    const float* __restrict__ bLG, const float* __restrict__ bRG,
    const float* __restrict__ bOG, bf16_t* __restrict__ Lws,
    bf16_t* __restrict__ Rws, float* __restrict__ Gws) {
  __shared__ bf16_t sXn[16 * DIMD];
  const int m0 = blockIdx.x * 16;
  const int tid = threadIdx.x;
  const int lane = tid & 31;
  const int wave = __builtin_amdgcn_readfirstlane(tid >> 5);   // scalar

  // --- LayerNorm: 16 lanes per point, 8 channels per lane ---
  {
    const int pt = 2 * wave + (lane >> 4);
    const int ch = (lane & 15) * 8;
    const float* xr = x + (size_t)(m0 + pt) * DIMD + ch;
    const float4 u0 = *(const float4*)(xr);
    const float4 u1 = *(const float4*)(xr + 4);
    float v[8] = {u0.x, u0.y, u0.z, u0.w, u1.x, u1.y, u1.z, u1.w};
    float s = 0.f, s2 = 0.f;
#pragma unroll
    for (int e = 0; e < 8; ++e) { s += v[e]; s2 += v[e] * v[e]; }
#pragma unroll
    for (int m = 8; m >= 1; m >>= 1) {
      s += __shfl_xor(s, m, 32);
      s2 += __shfl_xor(s2, m, 32);
    }
    const float mean = s * (1.0f / DIMD);
    const float var = s2 * (1.0f / DIMD) - mean * mean;
    const float rinv = rsqrtf(var + LNEPS);
#pragma unroll
    for (int e = 0; e < 8; ++e)
      sXn[pt * DIMD + ch + e] =
          to_bf16((v[e] - mean) * rinv * nw[ch + e] + nb[ch + e]);
  }
  __syncthreads();

  // Hoist the xn A-operands for all four K steps (shared by every phase).
  bf16x16 aop[4];
#pragma unroll
  for (int s = 0; s < 4; ++s) aop[s] = load_a(sXn, DIMD, lane, s * 32);

  const int h0 = wave * 16;
  const int hh = h0 + (lane & 15);
  const int r = m0 / SEQN;   // first pair index (k in the einsum)
  const int c0 = m0 % SEQN;  // second pair index base

  // Phases 0/1: projection * sigmoid(gate) -> pre-transposed bf16 plane.
  auto gated_pair = [&](const bf16_t* wp, const bf16_t* wg, const float* pbv,
                        const float* gbv, bf16_t* dst) {
    f32x8 accP = {}, accG = {};
#pragma unroll
    for (int s = 0; s < 4; ++s) {
      accP = wmma_bf16(aop[s], load_bt(wp + (size_t)h0 * DIMD, DIMD, lane, s * 32), accP);
      accG = wmma_bf16(aop[s], load_bt(wg + (size_t)h0 * DIMD, DIMD, lane, s * 32), accG);
    }
    const float pb = pbv[hh], gb = gbv[hh];
    const size_t base = (size_t)hh * NN + r;
#pragma unroll
    for (int v = 0; v < 8; ++v) {
      const int pt = v + ((lane < 16) ? 0 : 8);
      dst[base + (size_t)(c0 + pt) * SEQN] =
          to_bf16((accP[v] + pb) * sigm(accG[v] + gb));
    }
  };
  gated_pair(wB, wB + 2 * 16384, bL, bLG, Lws);                 // left
  gated_pair(wB + 16384, wB + 3 * 16384, bR, bRG, Rws);         // right

  // Phase 2: sigmoid(ogate) -> Gws[m][h] fp32.
  {
    const bf16_t* wOG = wB + 4 * 16384;
    f32x8 acc = {};
#pragma unroll
    for (int s = 0; s < 4; ++s)
      acc = wmma_bf16(aop[s], load_bt(wOG + (size_t)h0 * DIMD, DIMD, lane, s * 32), acc);
    const float ob = bOG[hh];
#pragma unroll
    for (int v = 0; v < 8; ++v) {
      const int pt = v + ((lane < 16) ? 0 : 8);
      Gws[(size_t)(m0 + pt) * HIDH + hh] = sigm(acc[v] + ob);
    }
  }
}

// ---------------------------------------------------------------------------
// Kernel 2: triangle mix.  out[i,j,h] = sum_k right[k,i,h]*left[k,j,h].
// Per channel h this is a 384x384x384 GEMM on bf16 planes (L2 resident).
// Wave 0 stages A (128x32) and B (64x32) K-slabs into double-buffered LDS
// with TDM tensor_load_to_lds; all 8 waves consume via ds_load_b128 and
// accumulate 32x32 fp32 tiles (4 WMMA per K step per wave).
// Grid: (h=128, i-slabs=3 of 128, j-slabs=6 of 64).
// ---------------------------------------------------------------------------
__global__ __launch_bounds__(256) void k_tri(const bf16_t* __restrict__ Lws,
                                             const bf16_t* __restrict__ Rws,
                                             float* __restrict__ Ows) {
  __shared__ bf16_t sA[2][128 * 32];   // 16 KB
  __shared__ bf16_t sB[2][64 * 32];    //  8 KB
  const int h = blockIdx.x;
  const int tid = threadIdx.x;
  const int lane = tid & 31;
  const int wave = __builtin_amdgcn_readfirstlane(tid >> 5);   // scalar
  const int i0wg = blockIdx.y * 128;
  const int j0wg = blockIdx.z * 64;
  const int i0l = (wave >> 1) * 32;
  const int j0l = (wave & 1) * 32;
  const bf16_t* Ag = Rws + (size_t)h * NN + (size_t)i0wg * SEQN;  // [i][k]
  const bf16_t* Bg = Lws + (size_t)h * NN + (size_t)j0wg * SEQN;  // [j][k]

  if (wave == 0) {
    tdm_load_2d(lds_off(&sA[0][0]), Ag, 128);
    tdm_load_2d(lds_off(&sB[0][0]), Bg, 64);
    __builtin_amdgcn_s_wait_tensorcnt(0);
  }
  __syncthreads();

  f32x8 acc[2][2] = {};
  int p = 0;
  for (int kb = 0; kb < SEQN; kb += 32) {
    const bool more = (kb + 32) < SEQN;
    if (wave == 0 && more) {   // prefetch next K slab into the other buffer
      tdm_load_2d(lds_off(&sA[p ^ 1][0]), Ag + kb + 32, 128);
      tdm_load_2d(lds_off(&sB[p ^ 1][0]), Bg + kb + 32, 64);
    }
    const bf16x16 a0 = load_a(&sA[p][0] + (size_t)i0l * 32, 32, lane, 0);
    const bf16x16 a1 = load_a(&sA[p][0] + (size_t)(i0l + 16) * 32, 32, lane, 0);
    const bf16x16 b0 = load_bt(&sB[p][0] + (size_t)j0l * 32, 32, lane, 0);
    const bf16x16 b1 = load_bt(&sB[p][0] + (size_t)(j0l + 16) * 32, 32, lane, 0);
    acc[0][0] = wmma_bf16(a0, b0, acc[0][0]);
    acc[0][1] = wmma_bf16(a0, b1, acc[0][1]);
    acc[1][0] = wmma_bf16(a1, b0, acc[1][0]);
    acc[1][1] = wmma_bf16(a1, b1, acc[1][1]);
    if (wave == 0 && more) __builtin_amdgcn_s_wait_tensorcnt(0);
    __syncthreads();
    p ^= 1;
  }

  float* O = Ows + (size_t)h * NN;   // [i][j] fp32, contiguous writes
#pragma unroll
  for (int mi = 0; mi < 2; ++mi)
#pragma unroll
    for (int nj = 0; nj < 2; ++nj)
#pragma unroll
      for (int v = 0; v < 8; ++v) {
        const int i = i0wg + i0l + mi * 16 + v + ((lane < 16) ? 0 : 8);
        const int j = j0wg + j0l + nj * 16 + (lane & 15);
        O[(size_t)i * SEQN + j] = acc[mi][nj][v];
      }
}

// ---------------------------------------------------------------------------
// Kernel 3: LayerNorm over H + output projection + out-gate for 16 points.
// Each of the 8 waves computes one 16x16 (points x d) tile, K = 128.
// ---------------------------------------------------------------------------
__global__ __launch_bounds__(256) void k_out(
    const float* __restrict__ Ows, const float* __restrict__ Gws,
    const float* __restrict__ tw, const float* __restrict__ tb,
    const bf16_t* __restrict__ wOut, const float* __restrict__ bOut,
    float* __restrict__ out) {
  __shared__ bf16_t sT[16 * HIDH];
  const int m0 = blockIdx.x * 16;
  const int r = m0 / SEQN, c0 = m0 % SEQN;
  const int tid = threadIdx.x;
  const int lane = tid & 31;
  const int wave = __builtin_amdgcn_readfirstlane(tid >> 5);   // scalar

  // --- gather mix rows from [h][i][j] and LayerNorm over h ---
  {
    const int pt = 2 * wave + (lane >> 4);
    const int hb = (lane & 15) * 8;
    const float* base = Ows + (size_t)r * SEQN + (c0 + pt);
    float v[8];
#pragma unroll
    for (int e = 0; e < 8; ++e) v[e] = base[(size_t)(hb + e) * NN];
    float s = 0.f, s2 = 0.f;
#pragma unroll
    for (int e = 0; e < 8; ++e) { s += v[e]; s2 += v[e] * v[e]; }
#pragma unroll
    for (int m = 8; m >= 1; m >>= 1) {
      s += __shfl_xor(s, m, 32);
      s2 += __shfl_xor(s2, m, 32);
    }
    const float mean = s * (1.0f / HIDH);
    const float var = s2 * (1.0f / HIDH) - mean * mean;
    const float rinv = rsqrtf(var + LNEPS);
#pragma unroll
    for (int e = 0; e < 8; ++e)
      sT[pt * HIDH + hb + e] =
          to_bf16((v[e] - mean) * rinv * tw[hb + e] + tb[hb + e]);
  }
  __syncthreads();

  bf16x16 aop[4];
#pragma unroll
  for (int s = 0; s < 4; ++s) aop[s] = load_a(sT, HIDH, lane, s * 32);

  const int d0 = wave * 16;
  f32x8 acc = {};
#pragma unroll
  for (int s = 0; s < 4; ++s)
    acc = wmma_bf16(aop[s], load_bt(wOut + (size_t)d0 * HIDH, HIDH, lane, s * 32), acc);

  const int dd = d0 + (lane & 15);
  const float ob = bOut[dd];
#pragma unroll
  for (int v = 0; v < 8; ++v) {
    const int pt = v + ((lane < 16) ? 0 : 8);
    const size_t idx = (size_t)(m0 + pt) * DIMD + dd;
    out[idx] = (acc[v] + ob) * Gws[idx];   // DIM == HID -> gate channel == d
  }
}

// ---------------------------------------------------------------------------
extern "C" void kernel_launch(void* const* d_in, const int* in_sizes, int n_in,
                              void* d_out, int out_size, void* d_ws,
                              size_t ws_size, hipStream_t stream) {
  (void)in_sizes; (void)n_in; (void)out_size; (void)ws_size;
  const float* x      = (const float*)d_in[0];
  const float* norm_w = (const float*)d_in[1];
  const float* norm_b = (const float*)d_in[2];
  const float* left_w = (const float*)d_in[3];
  const float* left_b = (const float*)d_in[4];
  const float* right_w = (const float*)d_in[5];
  const float* right_b = (const float*)d_in[6];
  const float* lgate_w = (const float*)d_in[7];
  const float* lgate_b = (const float*)d_in[8];
  const float* rgate_w = (const float*)d_in[9];
  const float* rgate_b = (const float*)d_in[10];
  const float* ogate_w = (const float*)d_in[11];
  const float* ogate_b = (const float*)d_in[12];
  const float* ton_w  = (const float*)d_in[13];
  const float* ton_b  = (const float*)d_in[14];
  const float* out_w  = (const float*)d_in[15];
  const float* out_b  = (const float*)d_in[16];
  float* out = (float*)d_out;

  char* ws = (char*)d_ws;
  const size_t offW = 0;                                   // 6*16384 bf16
  const size_t offL = offW + 6 * 16384 * sizeof(bf16_t);   // 196608 B
  const size_t offR = offL + (size_t)HIDH * NN * sizeof(bf16_t);
  const size_t offG = offR + (size_t)HIDH * NN * sizeof(bf16_t);
  const size_t offO = offG + (size_t)NN * HIDH * sizeof(float);
  bf16_t* wBf = (bf16_t*)(ws + offW);
  bf16_t* Lws = (bf16_t*)(ws + offL);
  bf16_t* Rws = (bf16_t*)(ws + offR);
  float*  Gws = (float*)(ws + offG);
  float*  Ows = (float*)(ws + offO);

  k_cvt<<<dim3(64, 6), 256, 0, stream>>>(left_w, right_w, lgate_w, rgate_w,
                                         ogate_w, out_w, wBf);
  k_proj<<<NN / 16, 256, 0, stream>>>(x, norm_w, norm_b, wBf, left_b, right_b,
                                      lgate_b, rgate_b, ogate_b, Lws, Rws, Gws);
  k_tri<<<dim3(HIDH, SEQN / 128, SEQN / 64), 256, 0, stream>>>(Lws, Rws, Ows);
  k_out<<<NN / 16, 256, 0, stream>>>(Ows, Gws, ton_w, ton_b,
                                     wBf + 5 * 16384, out_b, out);
}
